// MultiHeadAttention_19361712570503
// MI455X (gfx1250) — matrix operand order repaired and checked by reading, not compiled, not run on hardware
//
#include <hip/hip_runtime.h>
#include <hip/hip_bf16.h>
#include <math.h>

typedef __bf16 bf16;
typedef __bf16 v16bf __attribute__((ext_vector_type(16)));
typedef __bf16 v8bf  __attribute__((ext_vector_type(8)));
typedef float  v8f   __attribute__((ext_vector_type(8)));

#define BB 2
#define NN 2048
#define DD 1024
#define HH 16
#define HDD 64
#define ROTD 32

// ---- CDNA5 async global->LDS via inline asm (ISA 15.18.3, GV mode) --------
// vdst = LDS byte offset VGPR, vaddr = 64-bit global address, saddr = off.
// Tracked by ASYNCcnt; wave must s_wait_asynccnt 0 before LDS is readable.
#define ASYNC_COPY_B128(ldsoff, gaddr)                                        \
  asm volatile("global_load_async_to_lds_b128 %0, %1, off"                    \
               :: "v"(ldsoff), "v"(gaddr) : "memory")
#define WAIT_ASYNC() asm volatile("s_wait_asynccnt 0x0" ::: "memory")

__device__ __forceinline__ bf16 f2bf(float f) {
  unsigned u = __builtin_bit_cast(unsigned, f);
  unsigned r = (u + 0x7FFFu + ((u >> 16) & 1u)) >> 16;
  unsigned short s = (unsigned short)r;
  return __builtin_bit_cast(bf16, s);
}

union AFrag { v16bf v; v8bf h[2]; };

// ds_swizzle group-of-32: offset = xor_mask<<10 | or_mask<<5 | and_mask(0x1F)
#define SWZ_XOR(x, m)                                                         \
  __builtin_bit_cast(float, __builtin_amdgcn_ds_swizzle(                      \
      __builtin_bit_cast(int, (x)), (((m) << 10) | 0x1F)))

__device__ __forceinline__ float rowmax16(float x) {
  x = fmaxf(x, SWZ_XOR(x, 1));
  x = fmaxf(x, SWZ_XOR(x, 2));
  x = fmaxf(x, SWZ_XOR(x, 4));
  x = fmaxf(x, SWZ_XOR(x, 8));
  return x;
}
__device__ __forceinline__ float rowsum16(float x) {
  x += SWZ_XOR(x, 1);
  x += SWZ_XOR(x, 2);
  x += SWZ_XOR(x, 4);
  x += SWZ_XOR(x, 8);
  return x;
}

// ---------------------------------------------------------------------------
// Projection GEMM: Y[m][n] = sum_k X[m][k] * W[n][k]   (M=4096, N=1024, K=1024)
// MODE 0: Q proj  (f32 in) -> RoPE -> *0.125 -> bf16 ws (B,H,N,HD)
// MODE 1: K proj  (f32 in) -> RoPE -> bf16 ws (B,H,N,HD) + f32 k_t (B,H,HD,N)
// MODE 2: V proj  (f32 in) -> bf16 ws transposed (B,H,HD,N) + f32 vh (B,H,N,HD)
// MODE 3: O proj  (bf16 in, async LDS staging) -> f32 out (B,N,D)
// ---------------------------------------------------------------------------
template<int MODE>
__global__ __launch_bounds__(256)
void proj_kernel(const void* __restrict__ Xv, const float* __restrict__ W,
                 const float* __restrict__ rot,
                 bf16* __restrict__ bws, float* __restrict__ fout)
{
  __shared__ alignas(64) bf16 lA[64 * 32];
  __shared__ alignas(64) bf16 lB[64 * 32];

  const int tid  = threadIdx.x;
  const int wave = tid >> 5;
  const int lane = tid & 31;
  const int ln   = lane & 15;
  const int hi   = lane >> 4;
  const int wm   = wave & 3;   // m-tile within block (0..3)
  const int wn   = wave >> 2;  // n-group (0..1), 32 cols each
  const int mBase = blockIdx.y * 64;
  const int nBase = blockIdx.x * 64;

  v8f acc[2];
  #pragma unroll
  for (int t = 0; t < 2; ++t)
    #pragma unroll
    for (int i = 0; i < 8; ++i) acc[t][i] = 0.0f;

  const int srow = tid >> 2;        // 0..63
  const int scol = (tid & 3) * 8;   // 0,8,16,24

  for (int k0 = 0; k0 < DD; k0 += 32) {
    // ---- stage A tile (64 rows x 32 k) into LDS as bf16 ----
    if (MODE < 3) {
      const float* X = (const float*)Xv;
      const float* s = X + (size_t)(mBase + srow) * DD + k0 + scol;
      float4 f0 = *(const float4*)s;
      float4 f1 = *(const float4*)(s + 4);
      __builtin_prefetch(s + 32, 0, 0);  // next K tile
      bf16* d = &lA[srow * 32 + scol];
      d[0] = f2bf(f0.x); d[1] = f2bf(f0.y); d[2] = f2bf(f0.z); d[3] = f2bf(f0.w);
      d[4] = f2bf(f1.x); d[5] = f2bf(f1.y); d[6] = f2bf(f1.z); d[7] = f2bf(f1.w);
    } else {
      const bf16* X = (const bf16*)Xv;
      const bf16* s = X + (size_t)(mBase + srow) * DD + k0 + scol;
      // async DMA 16B straight into LDS (generic LDS ptr low 32 bits = offset)
      unsigned ldst = (unsigned)(size_t)&lA[srow * 32 + scol];
      ASYNC_COPY_B128(ldst, s);
    }
    // ---- stage B tile: B(k,n) = W[n][k], rows indexed by n ----
    {
      const float* s = W + (size_t)(nBase + srow) * DD + k0 + scol;
      float4 f0 = *(const float4*)s;
      float4 f1 = *(const float4*)(s + 4);
      __builtin_prefetch(s + 32, 0, 0);  // next K tile
      bf16* d = &lB[srow * 32 + scol];
      d[0] = f2bf(f0.x); d[1] = f2bf(f0.y); d[2] = f2bf(f0.z); d[3] = f2bf(f0.w);
      d[4] = f2bf(f1.x); d[5] = f2bf(f1.y); d[6] = f2bf(f1.z); d[7] = f2bf(f1.w);
    }
    if (MODE == 3) WAIT_ASYNC();
    __syncthreads();

    // A fragment: lane = row (m = ln), K chunks split across lane halves
    AFrag a;
    const bf16* ar = &lA[(wm * 16 + ln) * 32];
    a.h[0] = *(const v8bf*)(ar + hi * 16);
    a.h[1] = *(const v8bf*)(ar + hi * 16 + 8);

    #pragma unroll
    for (int t = 0; t < 2; ++t) {
      // B fragment: lane = column n, contiguous 16 k-values per lane-half
      const bf16* br = &lB[(wn * 32 + t * 16 + ln) * 32 + hi * 16];
      v16bf b = *(const v16bf*)br;
      acc[t] = __builtin_amdgcn_wmma_f32_16x16x32_bf16(
          false, a.v, false, b, (short)0, acc[t], false, false);
    }
    __syncthreads();
  }

  // ---- epilogue: RoPE (modes 0/1) + scattered layout stores ----
  #pragma unroll
  for (int t = 0; t < 2; ++t) {
    const int col = nBase + wn * 32 + t * 16 + ln;
    #pragma unroll
    for (int r = 0; r < 8; ++r) {
      const int row = mBase + wm * 16 + r + 8 * hi;
      float val = acc[t][r];

      if (MODE <= 1) {
        const int pos = row & (NN - 1);
        const int hd  = col & (HDD - 1);
        // neighbor column (col^1) lives in lane^1, same row slot
        float other = SWZ_XOR(val, 1);
        if (hd < ROTD) {  // uniform per 16-wide tile
          float f = rot[pos * ROTD + hd];
          float sn, cs;
          __sincosf(f, &sn, &cs);
          val = (hd & 1) ? fmaf(other, sn, val * cs)
                         : fmaf(-other, sn, val * cs);
        }
      }

      if (MODE == 0) {
        const int b = row >> 11, pos = row & (NN - 1);
        const int h = col >> 6,  hd  = col & (HDD - 1);
        // fold 1/sqrt(HD)=0.125 into Q (exact power-of-two scale)
        bws[((size_t)(b * HH + h) * NN + pos) * HDD + hd] = f2bf(val * 0.125f);
      } else if (MODE == 1) {
        const int b = row >> 11, pos = row & (NN - 1);
        const int h = col >> 6,  hd  = col & (HDD - 1);
        bws[((size_t)(b * HH + h) * NN + pos) * HDD + hd] = f2bf(val);
        fout[((size_t)(b * HH + h) * HDD + hd) * NN + pos] = val;   // k_t
      } else if (MODE == 2) {
        const int b = row >> 11, pos = row & (NN - 1);
        const int h = col >> 6,  hd  = col & (HDD - 1);
        bws[((size_t)(b * HH + h) * HDD + hd) * NN + pos] = f2bf(val); // V^T
        fout[((size_t)(b * HH + h) * NN + pos) * HDD + hd] = val;      // vh
      } else {
        fout[(size_t)row * DD + col] = val;
      }
    }
  }
}

// ---------------------------------------------------------------------------
// Flash attention: per (b,h), 64 q-rows per block (4 waves x 16 rows),
// online softmax over 64-key blocks (halves cross-lane reduction overhead).
// Q (pre-scaled by 0.125) / K in (B,H,N,HD) bf16, V in (B,H,HD,N) bf16.
// ---------------------------------------------------------------------------
__global__ __launch_bounds__(128)
void attn_kernel(const bf16* __restrict__ qw, const bf16* __restrict__ kw,
                 const bf16* __restrict__ vw, bf16* __restrict__ ow)
{
  __shared__ alignas(64) bf16 lP[4][16 * 64];

  const int tid  = threadIdx.x;
  const int wave = tid >> 5;
  const int lane = tid & 31;
  const int ln   = lane & 15;
  const int hi   = lane >> 4;
  const int bh   = blockIdx.y;           // b*16 + h
  const int q0   = blockIdx.x * 64 + wave * 16;
  const int b    = bh >> 4;
  const int h    = bh & 15;

  // preload Q A-fragments (HD chunks 0..31, 32..63)
  AFrag qf[2];
  {
    const bf16* qr = qw + ((size_t)bh * NN + q0 + ln) * HDD;
    #pragma unroll
    for (int c = 0; c < 2; ++c) {
      qf[c].h[0] = *(const v8bf*)(qr + c * 32 + hi * 16);
      qf[c].h[1] = *(const v8bf*)(qr + c * 32 + hi * 16 + 8);
    }
  }

  v8f o[4];
  #pragma unroll
  for (int j = 0; j < 4; ++j)
    #pragma unroll
    for (int i = 0; i < 8; ++i) o[j][i] = 0.0f;

  float mst[8], lst[8];
  #pragma unroll
  for (int r = 0; r < 8; ++r) { mst[r] = -INFINITY; lst[r] = 0.0f; }

  for (int key0 = 0; key0 < NN; key0 += 64) {
    // ---- S = (Q/8) K^T : four 16x16 key tiles, contraction over HD=64 ----
    v8f s[4];
    #pragma unroll
    for (int t = 0; t < 4; ++t) {
      #pragma unroll
      for (int i = 0; i < 8; ++i) s[t][i] = 0.0f;
      #pragma unroll
      for (int c = 0; c < 2; ++c) {
        const bf16* kr = kw + ((size_t)bh * NN + key0 + t * 16 + ln) * HDD
                            + c * 32 + hi * 16;
        v16bf kb = *(const v16bf*)kr;
        s[t] = __builtin_amdgcn_wmma_f32_16x16x32_bf16(
            false, qf[c].v, false, kb, (short)0, s[t], false, false);
      }
    }

    // ---- online softmax update (row r+8*hi spans the 16 lanes of a half) ----
    #pragma unroll
    for (int r = 0; r < 8; ++r) {
      float rm = fmaxf(fmaxf(s[0][r], s[1][r]), fmaxf(s[2][r], s[3][r]));
      rm = rowmax16(rm);
      float mnew  = fmaxf(mst[r], rm);
      float alpha = __expf(mst[r] - mnew);
      float p0 = __expf(s[0][r] - mnew);
      float p1 = __expf(s[1][r] - mnew);
      float p2 = __expf(s[2][r] - mnew);
      float p3 = __expf(s[3][r] - mnew);
      lst[r] = lst[r] * alpha + rowsum16((p0 + p1) + (p2 + p3));
      mst[r] = mnew;
      #pragma unroll
      for (int j = 0; j < 4; ++j) o[j][r] *= alpha;
      bf16* pd = &lP[wave][(r + 8 * hi) * 64 + ln];
      pd[0]  = f2bf(p0);
      pd[16] = f2bf(p1);
      pd[32] = f2bf(p2);
      pd[48] = f2bf(p3);
    }

    // ---- repack P (C-layout -> two A-fragments via per-wave LDS patch) ----
    AFrag pf[2];
    #pragma unroll
    for (int kc = 0; kc < 2; ++kc) {
      const bf16* pr = &lP[wave][ln * 64 + kc * 32];
      pf[kc].h[0] = *(const v8bf*)(pr + hi * 16);
      pf[kc].h[1] = *(const v8bf*)(pr + hi * 16 + 8);
    }

    // ---- O += P V  (V stored transposed: contiguous keys per lane) ----
    #pragma unroll
    for (int j = 0; j < 4; ++j) {
      #pragma unroll
      for (int kc = 0; kc < 2; ++kc) {
        const bf16* vr = vw + ((size_t)bh * HDD + j * 16 + ln) * NN
                            + key0 + kc * 32 + hi * 16;
        v16bf vb = *(const v16bf*)vr;
        o[j] = __builtin_amdgcn_wmma_f32_16x16x32_bf16(
            false, pf[kc].v, false, vb, (short)0, o[j], false, false);
      }
    }
  }

  // ---- normalize and write attention output (B,N,D) bf16 ----
  float invl[8];
  #pragma unroll
  for (int r = 0; r < 8; ++r) invl[r] = 1.0f / lst[r];
  #pragma unroll
  for (int j = 0; j < 4; ++j) {
    #pragma unroll
    for (int r = 0; r < 8; ++r) {
      const int qrow = q0 + r + 8 * hi;
      const int colD = h * HDD + j * 16 + ln;
      ow[((size_t)b * NN + qrow) * DD + colD] = f2bf(o[j][r] * invl[r]);
    }
  }
}

// ---------------------------------------------------------------------------
extern "C" void kernel_launch(void* const* d_in, const int* in_sizes, int n_in,
                              void* d_out, int out_size, void* d_ws, size_t ws_size,
                              hipStream_t stream)
{
  const float* q   = (const float*)d_in[0];
  const float* k   = (const float*)d_in[1];
  const float* v   = (const float*)d_in[2];
  const float* wq  = (const float*)d_in[3];
  const float* wk  = (const float*)d_in[4];
  const float* wv  = (const float*)d_in[5];
  const float* wo  = (const float*)d_in[6];
  const float* rot = (const float*)d_in[7];

  const size_t TOK = (size_t)BB * NN * DD;  // 4194304
  float* out = (float*)d_out;
  float* kt  = out + TOK;
  float* vh  = kt + TOK;

  bf16* qws = (bf16*)d_ws;
  bf16* kws = qws + TOK;
  bf16* vws = kws + TOK;
  bf16* ows = vws + TOK;

  dim3 gp(DD / 64, (BB * NN) / 64);  // (16, 64)
  dim3 bp(256);

  proj_kernel<0><<<gp, bp, 0, stream>>>(q, wq, rot, qws, nullptr);
  proj_kernel<1><<<gp, bp, 0, stream>>>(k, wk, rot, kws, kt);
  proj_kernel<2><<<gp, bp, 0, stream>>>(v, wv, rot, vws, vh);

  attn_kernel<<<dim3(NN / 64, BB * HH), dim3(128), 0, stream>>>(qws, kws, vws, ows);

  proj_kernel<3><<<gp, bp, 0, stream>>>(ows, wo, nullptr, nullptr, out);
}
